// MLA_47708496724492
// MI455X (gfx1250) — matrix operand order repaired and checked by reading, not compiled, not run on hardware
//
#include <hip/hip_runtime.h>
#include <hip/hip_bf16.h>

// ---------------------------------------------------------------------------
// MLA forward for MI455X (gfx1250): all matmuls via v_wmma_f32_16x16x32_bf16.
// GEMM: 128x128 block tile, BK=32, 8 waves (4Mx2N), 8 WMMA/wave/k-step,
// register-staged global->LDS pipeline, vectorized (b128) LDS staging.
// Attention: 64-query causal flash blocks with online softmax in WMMA C layout.
// ---------------------------------------------------------------------------

typedef __attribute__((ext_vector_type(16))) __bf16 v16bf;
typedef __attribute__((ext_vector_type(8)))  __bf16 v8bf;
typedef __attribute__((ext_vector_type(8)))  float  v8f;

#define WMMA_BF16(a, b, c) \
  __builtin_amdgcn_wmma_f32_16x16x32_bf16(false, (a), false, (b), (short)0, (c), false, false)

// ---- WMMA fragment loaders (contiguous k = hi*16 + e in A and B, wave32) ----------

// A (16x32 bf16, MxK): lane m=lane&15, half hi=lane>>4.
// elem e -> K = hi*16 + e (contiguous, matches B loaders)
__device__ static inline v16bf load_A_frag(const __bf16* lds, int row0, int ld, int k0) {
  int lane = threadIdx.x & 31;
  int m = lane & 15, hi = lane >> 4;
  const __bf16* p = lds + (size_t)(row0 + m) * ld + k0;
  v16bf a;
#pragma unroll
  for (int e = 0; e < 16; ++e) a[e] = p[hi * 16 + e];
  return a;
}

// B (32x16 bf16, KxN): lane n=lane&15, hi=lane>>4; elem e -> K = hi*16 + e.
// LDS holds B transposed (col-major: lds[n*ld + k]) -> contiguous in e.
__device__ static inline v16bf load_B_frag_T(const __bf16* lds, int n0, int ld, int k0) {
  int lane = threadIdx.x & 31;
  int n = lane & 15, hi = lane >> 4;
  const __bf16* p = lds + (size_t)(n0 + n) * ld + k0 + hi * 16;
  v16bf b;
#pragma unroll
  for (int e = 0; e < 16; ++e) b[e] = p[e];
  return b;
}

// B from row-major LDS (lds[k*ld + n]) -- used for P @ V (V is key-major).
__device__ static inline v16bf load_B_frag_R(const __bf16* lds, int k0, int ld, int n0) {
  int lane = threadIdx.x & 31;
  int n = lane & 15, hi = lane >> 4;
  v16bf b;
#pragma unroll
  for (int e = 0; e < 16; ++e) b[e] = lds[(size_t)(k0 + hi * 16 + e) * ld + n0 + n];
  return b;
}

// pack 8 floats -> v8bf (v_cvt_pk_bf16_f32 pairs) for one ds_store_b128
__device__ static inline v8bf pack8(const float* f) {
  v8bf p;
#pragma unroll
  for (int e = 0; e < 8; ++e) p[e] = (__bf16)f[e];
  return p;
}

// ---------------------------------------------------------------------------
// GEMM: C[M,N] = A[M,K] * B[K,N], fp32 in/out, bf16 WMMA inside.
// Block tile 128x128, BK=32, 256 threads = 8 waves (4M x 2N), each wave a
// 32x64 sub-tile (2x4 fragments, 8 WMMAs/k-step). Software-pipelined loads.
// Requires: M % 128 == 0, K % 32 == 0. N guarded.
// ---------------------------------------------------------------------------
__global__ __launch_bounds__(256) void mla_gemm_bf16(
    const float* __restrict__ A, const float* __restrict__ B, float* __restrict__ C,
    int M, int N, int K, int lda, int ldb, int ldc) {
  __shared__ __bf16 lA[128 * 32];   // row-major [m][k]
  __shared__ __bf16 lBt[128 * 32];  // transposed [n][k]

  int t = threadIdx.x;
  int lane = t & 31, w = t >> 5;
  int wm = w & 3, wn = w >> 2;           // 4(M) x 2(N) wave grid
  int bm = blockIdx.y * 128, bn = blockIdx.x * 128;

  v8f acc[2][4] = {};

  // Each thread owns 16 *contiguous* LDS elements of each tile:
  //   A: row arow = t>>1, k range [acol, acol+16)        (contiguous in LDS & global)
  //   Bt: col bnc = t>>1, k range [bkg, bkg+16)          (contiguous in LDS; global
  //       loads form two 64B lane-consecutive segments per wave)
  int arow = t >> 1, acol = (t & 1) * 16;
  int bnc = t >> 1, bkg = (t & 1) * 16;
  bool bok = (bn + bnc) < N;

  float ra[16], rb[16];

  auto issue_loads = [&](int k0) {
#pragma unroll
    for (int j = 0; j < 16; ++j)
      ra[j] = A[(size_t)(bm + arow) * lda + k0 + acol + j];
#pragma unroll
    for (int j = 0; j < 16; ++j)
      rb[j] = bok ? B[(size_t)(k0 + bkg + j) * ldb + bn + bnc] : 0.f;
  };

  issue_loads(0);

  for (int k0 = 0; k0 < K; k0 += 32) {
    // stage current tile into LDS: 2x ds_store_b128 per matrix per thread
    *(v8bf*)&lA[arow * 32 + acol]      = pack8(ra);
    *(v8bf*)&lA[arow * 32 + acol + 8]  = pack8(ra + 8);
    *(v8bf*)&lBt[bnc * 32 + bkg]       = pack8(rb);
    *(v8bf*)&lBt[bnc * 32 + bkg + 8]   = pack8(rb + 8);
    __syncthreads();

    // pipeline: kick off next tile's global loads; prefetch the tile after
    if (k0 + 32 < K) {
      issue_loads(k0 + 32);
      if (k0 + 64 < K) {
        __builtin_prefetch(&A[(size_t)(bm + arow) * lda + k0 + 64 + acol], 0, 3);
        __builtin_prefetch(&B[(size_t)(k0 + 64 + bkg) * ldb + bn + bnc], 0, 3);
      }
    }

    // compute: 2 A-frags x 4 B-frags -> 8 WMMAs
    v16bf a0 = load_A_frag(lA, wm * 32, 32, 0);
    v16bf a1 = load_A_frag(lA, wm * 32 + 16, 32, 0);
#pragma unroll
    for (int j = 0; j < 4; ++j) {
      v16bf bf = load_B_frag_T(lBt, wn * 64 + j * 16, 32, 0);
      acc[0][j] = WMMA_BF16(a0, bf, acc[0][j]);
      acc[1][j] = WMMA_BF16(a1, bf, acc[1][j]);
    }
    __syncthreads();
  }

  int hi = lane >> 4, ln = lane & 15;
#pragma unroll
  for (int i = 0; i < 2; ++i) {
#pragma unroll
    for (int j = 0; j < 4; ++j) {
#pragma unroll
      for (int r = 0; r < 8; ++r) {
        int row = bm + wm * 32 + i * 16 + r + hi * 8;
        int col = bn + wn * 64 + j * 16 + ln;
        if (col < N) C[(size_t)row * ldc + col] = acc[i][j][r];
      }
    }
  }
}

// ---------------------------------------------------------------------------
// In-place RMSNorm: one block per row.
// ---------------------------------------------------------------------------
__global__ __launch_bounds__(256) void mla_rmsnorm(float* __restrict__ x,
                                                   const float* __restrict__ g,
                                                   int cols, int ld) {
  __shared__ float red[256];
  float* p = x + (size_t)blockIdx.x * ld;
  int t = threadIdx.x;
  float ss = 0.f;
  for (int i = t; i < cols; i += 256) { float v = p[i]; ss += v * v; }
  red[t] = ss;
  __syncthreads();
  for (int s = 128; s > 0; s >>= 1) {
    if (t < s) red[t] += red[t + s];
    __syncthreads();
  }
  float scale = rsqrtf(red[0] / (float)cols + 1.1920929e-07f);
  for (int i = t; i < cols; i += 256) p[i] = p[i] * scale * g[i];
}

// ---------------------------------------------------------------------------
// RoPE for shared k_rope: kv_c[:, 512:576] -> krope[S,64]. 32 threads/row.
// rope_freq layout: [S][32][2] = cos,sin.
// ---------------------------------------------------------------------------
__global__ __launch_bounds__(32) void mla_rope_k(const float* __restrict__ kv_c,
                                                 const float* __restrict__ rf,
                                                 float* __restrict__ krope) {
  int s = blockIdx.x, i = threadIdx.x;  // i = pair index 0..31
  const float* src = kv_c + (size_t)s * 576 + 512;
  float xr = src[2 * i], xi = src[2 * i + 1];
  float c = rf[(size_t)s * 64 + 2 * i], sn = rf[(size_t)s * 64 + 2 * i + 1];
  krope[(size_t)s * 64 + 2 * i]     = xr * c - xi * sn;
  krope[(size_t)s * 64 + 2 * i + 1] = xr * sn + xi * c;
}

// In-place RoPE on q's rope tail: q[S, H*192], cols [h*192+128, h*192+192).
__global__ __launch_bounds__(32) void mla_rope_q(float* __restrict__ q,
                                                 const float* __restrict__ rf) {
  int s = blockIdx.x, h = blockIdx.y, i = threadIdx.x;
  float* p = q + (size_t)s * 3072 + h * 192 + 128;
  float xr = p[2 * i], xi = p[2 * i + 1];
  float c = rf[(size_t)s * 64 + 2 * i], sn = rf[(size_t)s * 64 + 2 * i + 1];
  p[2 * i]     = xr * c - xi * sn;
  p[2 * i + 1] = xr * sn + xi * c;
}

// ---------------------------------------------------------------------------
// Flash attention per head. Block = 64 queries x 1 head, 4 waves x 16 queries.
// K-dim of QK^T = 192 (nope128 + rope64); V dim = 128. Causal, online softmax.
// ---------------------------------------------------------------------------
__global__ __launch_bounds__(128) void mla_flash_attn(
    const float* __restrict__ q, const float* __restrict__ kv,
    const float* __restrict__ krope, float* __restrict__ attn_out) {
  __shared__ __bf16 lQ[64 * 192];       // [query][d]
  __shared__ __bf16 lK[32 * 192];       // [key][d]   (== B^T for Q.K^T)
  __shared__ __bf16 lV[32 * 128];       // [key][d]   (== row-major B for P.V)
  __shared__ __bf16 lP[4 * 16 * 32];    // per-wave P tile, 16x32

  const float SCALE = 0.07216878364870323f;  // 192^-0.5

  int h = blockIdx.y;
  int qbase = blockIdx.x * 64;
  int t = threadIdx.x, lane = t & 31, w = t >> 5;
  int hi = lane >> 4, ln = lane & 15;
  int m0 = qbase + w * 16;

  // Stage Q in 8-element contiguous chunks (192 % 8 == 0 -> chunk stays in row)
#pragma unroll
  for (int c = 0; c < 12; ++c) {
    int e0 = (t + 128 * c) * 8;
    int row = e0 / 192, col = e0 % 192;
    float f[8];
#pragma unroll
    for (int e = 0; e < 8; ++e)
      f[e] = q[(size_t)(qbase + row) * 3072 + h * 192 + col + e];
    *(v8bf*)&lQ[e0] = pack8(f);
  }

  v8f o[8] = {};
  float mrow[8], lrow[8];
#pragma unroll
  for (int i = 0; i < 8; ++i) { mrow[i] = -1e30f; lrow[i] = 0.f; }

  int nkt = qbase / 32 + 2;  // enough key tiles to cover causal range of block
  for (int kt = 0; kt < nkt; ++kt) {
    int key0 = kt * 32;
    __syncthreads();
    // Stage K = [k_nope | k_rope]; 128-col boundary is 8-aligned
#pragma unroll
    for (int c = 0; c < 6; ++c) {
      int e0 = (t + 128 * c) * 8;
      int row = e0 / 192, col = e0 % 192;
      int key = key0 + row;
      float f[8];
      if (col < 128) {
#pragma unroll
        for (int e = 0; e < 8; ++e) f[e] = kv[(size_t)key * 4096 + h * 256 + col + e];
      } else {
#pragma unroll
        for (int e = 0; e < 8; ++e) f[e] = krope[(size_t)key * 64 + col - 128 + e];
      }
      *(v8bf*)&lK[e0] = pack8(f);
    }
    // Stage V
#pragma unroll
    for (int c = 0; c < 4; ++c) {
      int e0 = (t + 128 * c) * 8;
      int row = e0 >> 7, col = e0 & 127;
      float f[8];
#pragma unroll
      for (int e = 0; e < 8; ++e)
        f[e] = kv[(size_t)(key0 + row) * 4096 + h * 256 + 128 + col + e];
      *(v8bf*)&lV[e0] = pack8(f);
    }
    __syncthreads();

    // S = Q . K^T  (16 queries x 32 keys): 2 accumulators x 6 k-steps
    v8f s0 = {}, s1 = {};
#pragma unroll
    for (int ks = 0; ks < 6; ++ks) {
      v16bf aq = load_A_frag(lQ, w * 16, 192, ks * 32);
      v16bf b0 = load_B_frag_T(lK, 0, 192, ks * 32);
      v16bf b1 = load_B_frag_T(lK, 16, 192, ks * 32);
      s0 = WMMA_BF16(aq, b0, s0);
      s1 = WMMA_BF16(aq, b1, s1);
    }

    // Online softmax in C layout: element r -> row (r + 8*hi), col (ln + {0,16})
    __bf16* Pw = lP + w * 512;
#pragma unroll
    for (int r = 0; r < 8; ++r) {
      int qg = m0 + r + hi * 8;
      float v0 = s0[r] * SCALE; if (key0 + ln > qg)      v0 = -1e30f;
      float v1 = s1[r] * SCALE; if (key0 + 16 + ln > qg) v1 = -1e30f;
      float tm = fmaxf(v0, v1);
      tm = fmaxf(tm, __shfl_xor(tm, 1, 32));
      tm = fmaxf(tm, __shfl_xor(tm, 2, 32));
      tm = fmaxf(tm, __shfl_xor(tm, 4, 32));
      tm = fmaxf(tm, __shfl_xor(tm, 8, 32));  // reduce within 16-lane half
      float mnew = fmaxf(mrow[r], tm);
      float p0 = __expf(v0 - mnew), p1 = __expf(v1 - mnew);
      float corr = __expf(mrow[r] - mnew);
      float ps = p0 + p1;
      ps += __shfl_xor(ps, 1, 32);
      ps += __shfl_xor(ps, 2, 32);
      ps += __shfl_xor(ps, 4, 32);
      ps += __shfl_xor(ps, 8, 32);
      lrow[r] = lrow[r] * corr + ps;
      mrow[r] = mnew;
#pragma unroll
      for (int nt = 0; nt < 8; ++nt) o[nt][r] *= corr;
      Pw[(r + hi * 8) * 32 + ln]      = (__bf16)p0;
      Pw[(r + hi * 8) * 32 + 16 + ln] = (__bf16)p1;
    }
    // Per-wave LDS round trip: make P stores visible to our own fragment loads.
    asm volatile("s_wait_dscnt 0" ::: "memory");

    // O += P . V   (one 16x16x32 WMMA per 16-wide V column tile)
    v16bf ap = load_A_frag(Pw, 0, 32, 0);
#pragma unroll
    for (int nt = 0; nt < 8; ++nt) {
      v16bf bv = load_B_frag_R(lV, 0, 128, nt * 16);
      o[nt] = WMMA_BF16(ap, bv, o[nt]);
    }
  }

  // Epilogue: attn_out[q][h*128 + d] = o / l
#pragma unroll
  for (int nt = 0; nt < 8; ++nt) {
#pragma unroll
    for (int r = 0; r < 8; ++r) {
      int qg = m0 + r + hi * 8;
      attn_out[(size_t)qg * 2048 + h * 128 + nt * 16 + ln] = o[nt][r] / lrow[r];
    }
  }
}

// ---------------------------------------------------------------------------
// Host-side orchestration.
// ---------------------------------------------------------------------------
extern "C" void kernel_launch(void* const* d_in, const int* in_sizes, int n_in,
                              void* d_out, int out_size, void* d_ws, size_t ws_size,
                              hipStream_t stream) {
  (void)in_sizes; (void)n_in; (void)out_size; (void)ws_size;
  const float* x     = (const float*)d_in[0];
  const float* rf    = (const float*)d_in[1];
  // d_in[2] = mask (unused; causal mask computed analytically)
  const float* wq_d  = (const float*)d_in[3];
  const float* q_g   = (const float*)d_in[4];
  const float* wq_u  = (const float*)d_in[5];
  const float* wkv_d = (const float*)d_in[6];
  const float* kv_g  = (const float*)d_in[7];
  const float* wkv_u = (const float*)d_in[8];
  const float* wo    = (const float*)d_in[9];
  float* out = (float*)d_out;

  const int S = 2048, D = 2048;

  // Workspace layout (fp32): ~87 MB total
  float* q_lat = (float*)d_ws;                 // [S,768]
  float* kv_c  = q_lat + (size_t)S * 768;      // [S,576]
  float* krope = kv_c + (size_t)S * 576;       // [S,64]
  float* qbuf  = krope + (size_t)S * 64;       // [S,3072]
  float* kvbuf = qbuf + (size_t)S * 3072;      // [S,4096]
  float* attn  = kvbuf + (size_t)S * 4096;     // [S,2048]

  dim3 blk256(256);
  auto gx = [](int n) { return (n + 127) / 128; };

  // 1) down-projections
  mla_gemm_bf16<<<dim3(gx(768), S / 128), blk256, 0, stream>>>(
      x, wq_d, q_lat, S, 768, D, D, 768, 768);
  mla_gemm_bf16<<<dim3(gx(576), S / 128), blk256, 0, stream>>>(
      x, wkv_d, kv_c, S, 576, D, D, 576, 576);

  // 2) rmsnorms (in place)
  mla_rmsnorm<<<S, blk256, 0, stream>>>(q_lat, q_g, 768, 768);
  mla_rmsnorm<<<S, blk256, 0, stream>>>(kv_c, kv_g, 512, 576);

  // 3) shared k_rope rotation (reads kv_c cols 512..575, untouched by rmsnorm)
  mla_rope_k<<<S, dim3(32), 0, stream>>>(kv_c, rf, krope);

  // 4) up-projections
  mla_gemm_bf16<<<dim3(gx(3072), S / 128), blk256, 0, stream>>>(
      q_lat, wq_u, qbuf, S, 3072, 768, 768, 3072, 3072);
  mla_rope_q<<<dim3(S, 16), dim3(32), 0, stream>>>(qbuf, rf);
  mla_gemm_bf16<<<dim3(gx(4096), S / 128), blk256, 0, stream>>>(
      kv_c, wkv_u, kvbuf, S, 4096, 512, 576, 4096, 4096);

  // 5) causal flash attention, 16 heads
  mla_flash_attn<<<dim3(S / 64, 16), dim3(128), 0, stream>>>(qbuf, kvbuf, krope, attn);

  // 6) output projection
  mla_gemm_bf16<<<dim3(gx(D), S / 128), blk256, 0, stream>>>(
      attn, wo, out, S, D, 2048, 2048, D, D);
}
